// FasterRCNN_VGG16_5815385718820
// MI455X (gfx1250) — matrix operand order repaired and checked by reading, not compiled, not run on hardware
//
#include <hip/hip_runtime.h>
#include <hip/hip_bf16.h>
#include <math.h>

typedef __attribute__((ext_vector_type(16))) _Float16 v16h;
typedef __attribute__((ext_vector_type(8)))  float    v8f;

#define NSORT   32768
#define PRE_NMS 6000
#define POST_N  300
#define FH 50
#define FW 38
#define NA (FH*FW*9)   // 17100
#define POOLC 512
#define IMGH 800
#define IMGW 608

#define AS1 __attribute__((address_space(1)))
#define AS3 __attribute__((address_space(3)))

#if __has_builtin(__builtin_amdgcn_global_load_async_to_lds_b32) && \
    __has_builtin(__builtin_amdgcn_s_wait_asynccnt)
#define USE_ASYNC_LDS 1
#else
#define USE_ASYNC_LDS 0
#endif

__device__ __forceinline__ void lds_copy_f32(const float* g, float* l)
{
#if USE_ASYNC_LDS
    __builtin_amdgcn_global_load_async_to_lds_b32((AS1 int*)g, (AS3 int*)l, 0, 0);
#else
    *l = *g;
#endif
}

__device__ __forceinline__ void lds_copy_wait()
{
#if USE_ASYNC_LDS
    __builtin_amdgcn_s_wait_asynccnt(0);
#endif
}

// Build A fragment (16x32 f16, ISA layout) from an LDS-resident 16x32 f32 chunk.
// rowValid==false -> zero fragment. kRem = K - k0 (>=32 means full chunk valid).
__device__ __forceinline__ v16h build_a_lds(const float* sA, int lo, int hi,
                                            bool rowValid, int kRem)
{
    v16h a = {};
    if (!rowValid) return a;
    if (kRem >= 32) {
        const float4* p0 = (const float4*)(sA + lo * 32 + (hi << 3));
        const float4* p1 = (const float4*)(sA + lo * 32 + 16 + (hi << 3));
        float4 x0 = p0[0], x1 = p0[1], x2 = p1[0], x3 = p1[1];
        a[0]=(_Float16)x0.x;  a[1]=(_Float16)x0.y;  a[2]=(_Float16)x0.z;  a[3]=(_Float16)x0.w;
        a[4]=(_Float16)x1.x;  a[5]=(_Float16)x1.y;  a[6]=(_Float16)x1.z;  a[7]=(_Float16)x1.w;
        a[8]=(_Float16)x2.x;  a[9]=(_Float16)x2.y;  a[10]=(_Float16)x2.z; a[11]=(_Float16)x2.w;
        a[12]=(_Float16)x3.x; a[13]=(_Float16)x3.y; a[14]=(_Float16)x3.z; a[15]=(_Float16)x3.w;
    } else {
#pragma unroll
        for (int i = 0; i < 16; ++i) {
            int k = ((i >> 3) << 4) + (hi << 3) + (i & 7);
            a[i] = (k < kRem) ? (_Float16)sA[lo * 32 + k] : (_Float16)0.f;
        }
    }
    return a;
}

// ---------------------------------------------------------------------------
// WMMA direct convolution. Block = 8 waves sharing one 16-OC weight chunk in
// LDS (async-filled); wave w handles pixel-tile (pixBlock*8 + w).
// ---------------------------------------------------------------------------
__global__ void conv_wmma(const float* __restrict__ in, const float* __restrict__ w,
                          const float* __restrict__ bias, float* __restrict__ out,
                          int Cin, int Cout, int H, int W, int ksize, int relu)
{
    __shared__ float sA[16 * 32];
    const int lane = threadIdx.x & 31;
    const int wave = threadIdx.x >> 5;
    const int HW = H * W;
    const int pixT = (HW + 15) >> 4;
    const int pixBlocks = (pixT + 7) >> 3;
    const int tOc = blockIdx.x / pixBlocks;
    const int pb  = blockIdx.x - tOc * pixBlocks;
    const int tPx = pb * 8 + wave;
    const int lo = lane & 15, hi = lane >> 4;
    const int K = Cin * ksize * ksize;

    const int m = tOc * 16 + lo;          // output channel row for A frag
    const int p = tPx * 16 + lo;          // pixel column for B frag
    const int oy = p / W, ox = p % W;

    v8f acc = {};
    for (int k0 = 0; k0 < K; k0 += 32) {
        // stage weight chunk [16 oc x 32 k] into LDS (all 8 waves cooperate)
        for (int e = threadIdx.x; e < 512; e += blockDim.x) {
            int row = e >> 5, kk = e & 31;
            int gm = tOc * 16 + row; if (gm >= Cout) gm = Cout - 1;
            int gk = k0 + kk;        if (gk >= K)    gk = K - 1;
            lds_copy_f32(&w[(size_t)gm * K + gk], &sA[e]);
        }
        lds_copy_wait();
        __syncthreads();

        v16h a = build_a_lds(sA, lo, hi, m < Cout, K - k0);

        v16h b;
#pragma unroll
        for (int i = 0; i < 16; ++i) {   // B: 32x16 f16, K = (i>>3)*16+hi*8+(i&7), N=lo
            int k = k0 + ((i >> 3) << 4) + (hi << 3) + (i & 7);
            float bv = 0.f;
            if (k < K && p < HW) {
                int ci, iy, ix;
                if (ksize == 3) {
                    ci = k / 9; int r = k - ci * 9;
                    int ky = r / 3, kx = r - ky * 3;
                    iy = oy + ky - 1; ix = ox + kx - 1;
                } else { ci = k; iy = oy; ix = ox; }
                if ((unsigned)iy < (unsigned)H && (unsigned)ix < (unsigned)W)
                    bv = in[((size_t)ci * H + iy) * W + ix];
            }
            b[i] = (_Float16)bv;
        }
        acc = __builtin_amdgcn_wmma_f32_16x16x32_f16(false, a, false, b,
                                                     (short)0, acc, false, false);
        __syncthreads();   // protect sA before next refill
    }
#pragma unroll
    for (int r = 0; r < 8; ++r) {        // C/D: VGPR r -> M = hi*8+r, N = lo
        int mm = tOc * 16 + (hi << 3) + r;
        if (mm < Cout && p < HW) {
            float v = acc[r] + bias[mm];
            if (relu) v = fmaxf(v, 0.f);
            out[(size_t)mm * HW + p] = v;
        }
    }
}

// ---------------------------------------------------------------------------
// WMMA GEMM: out[M,N] = A[M,K] * Wt[N,K]^T + bias[N].  K must be mult. of 32.
// Block = 8 waves sharing the A chunk (16 x 32) via async LDS; each wave owns
// one 16-wide N tile; weights read as float4 (global_load_b128).
// ---------------------------------------------------------------------------
__global__ void gemm_wmma(const float* __restrict__ A, const float* __restrict__ Wt,
                          const float* __restrict__ bias, float* __restrict__ out,
                          int M, int N, int K, int relu)
{
    __shared__ float sA[16 * 32];
    const int lane = threadIdx.x & 31;
    const int wave = threadIdx.x >> 5;
    const int nT = (N + 15) >> 4;
    const int nBlk = (nT + 7) >> 3;
    const int tm = blockIdx.x / nBlk;
    const int tn = (blockIdx.x - tm * nBlk) * 8 + wave;
    const int lo = lane & 15, hi = lane >> 4;
    const int m = tm * 16 + lo;
    const int n = tn * 16 + lo;

    v8f acc = {};
    for (int k0 = 0; k0 < K; k0 += 32) {
        for (int e = threadIdx.x; e < 512; e += blockDim.x) {
            int row = e >> 5, kk = e & 31;
            int gm = tm * 16 + row; if (gm >= M) gm = M - 1;
            lds_copy_f32(&A[(size_t)gm * K + k0 + kk], &sA[e]);
        }
        lds_copy_wait();
        __syncthreads();

        v16h a = build_a_lds(sA, lo, hi, m < M, K - k0);

        v16h b = {};
        if (n < N) {
            const float4* wp = (const float4*)(Wt + (size_t)n * K + k0 + (hi << 3));
            float4 w0 = wp[0], w1 = wp[1], w2 = wp[4], w3 = wp[5];
            b[0]=(_Float16)w0.x;  b[1]=(_Float16)w0.y;  b[2]=(_Float16)w0.z;  b[3]=(_Float16)w0.w;
            b[4]=(_Float16)w1.x;  b[5]=(_Float16)w1.y;  b[6]=(_Float16)w1.z;  b[7]=(_Float16)w1.w;
            b[8]=(_Float16)w2.x;  b[9]=(_Float16)w2.y;  b[10]=(_Float16)w2.z; b[11]=(_Float16)w2.w;
            b[12]=(_Float16)w3.x; b[13]=(_Float16)w3.y; b[14]=(_Float16)w3.z; b[15]=(_Float16)w3.w;
            if (k0 + 32 < K)
                __builtin_prefetch(&Wt[(size_t)n * K + k0 + 32], 0, 0);
        }
        acc = __builtin_amdgcn_wmma_f32_16x16x32_f16(false, a, false, b,
                                                     (short)0, acc, false, false);
        __syncthreads();
    }
#pragma unroll
    for (int r = 0; r < 8; ++r) {
        int mm = tm * 16 + (hi << 3) + r;
        if (mm < M && n < N) {
            float v = acc[r] + bias[n];
            if (relu) v = fmaxf(v, 0.f);
            out[(size_t)mm * N + n] = v;
        }
    }
}

// ---------------------------------------------------------------------------
__global__ void maxpool_k(const float* __restrict__ in, float* __restrict__ out,
                          int C, int H, int W)
{
    int Ho = H >> 1, Wo = W >> 1;
    int n = C * Ho * Wo;
    int t = blockIdx.x * blockDim.x + threadIdx.x;
    if (t >= n) return;
    int c = t / (Ho * Wo);
    int r = t - c * (Ho * Wo);
    int y = r / Wo, x = r - y * Wo;
    const float* p = in + ((size_t)c * H + 2 * y) * W + 2 * x;
    out[t] = fmaxf(fmaxf(p[0], p[1]), fmaxf(p[W], p[W + 1]));
}

// ---------------------------------------------------------------------------
__global__ void proposal_prep(const float* __restrict__ loc, const float* __restrict__ sc,
                              const float* __restrict__ im_info,
                              float* __restrict__ props, float* __restrict__ keys,
                              int* __restrict__ idxs)
{
    const float base[9][4] = {
        { -84.f,  -40.f,  99.f,  55.f}, {-176.f,  -88.f, 191.f, 103.f},
        {-360.f, -184.f, 375.f, 199.f}, { -56.f,  -56.f,  71.f,  71.f},
        {-120.f, -120.f, 135.f, 135.f}, {-248.f, -248.f, 263.f, 263.f},
        { -36.f,  -80.f,  51.f,  95.f}, { -80.f, -168.f,  95.f, 183.f},
        {-168.f, -344.f, 183.f, 359.f}};
    int i = blockIdx.x * blockDim.x + threadIdx.x;
    if (i >= NSORT) return;
    if (i >= NA) { keys[i] = -INFINITY; idxs[i] = i; return; }
    int p = i / 9, a = i - 9 * p;
    int y = p / FW, x = p - y * FW;
    const int HW = FH * FW;
    float ax1 = base[a][0] + x * 16.f, ay1 = base[a][1] + y * 16.f;
    float ax2 = base[a][2] + x * 16.f, ay2 = base[a][3] + y * 16.f;
    float w = ax2 - ax1 + 1.f, h = ay2 - ay1 + 1.f;
    float cx = ax1 + 0.5f * w, cy = ay1 + 0.5f * h;
    float dx = loc[(size_t)(4 * a + 0) * HW + p], dy = loc[(size_t)(4 * a + 1) * HW + p];
    float dw = loc[(size_t)(4 * a + 2) * HW + p], dh = loc[(size_t)(4 * a + 3) * HW + p];
    float pcx = dx * w + cx, pcy = dy * h + cy;
    float pw = expf(dw) * w, ph = expf(dh) * h;
    float x1 = pcx - 0.5f * pw, y1 = pcy - 0.5f * ph;
    float x2 = pcx + 0.5f * pw, y2 = pcy + 0.5f * ph;
    float imH = im_info[0], imW = im_info[1], scale = im_info[2];
    x1 = fminf(fmaxf(x1, 0.f), imW - 1.f);  y1 = fminf(fmaxf(y1, 0.f), imH - 1.f);
    x2 = fminf(fmaxf(x2, 0.f), imW - 1.f);  y2 = fminf(fmaxf(y2, 0.f), imH - 1.f);
    props[i * 4 + 0] = x1; props[i * 4 + 1] = y1;
    props[i * 4 + 2] = x2; props[i * 4 + 3] = y2;
    float ms = 16.f * scale;
    bool valid = (x2 - x1 + 1.f >= ms) && (y2 - y1 + 1.f >= ms);
    float s1 = sc[(size_t)(2 * a) * HW + p], s2 = sc[(size_t)(2 * a + 1) * HW + p];
    float mx = fmaxf(s1, s2);
    float e1 = expf(s1 - mx), e2 = expf(s2 - mx);
    keys[i] = valid ? (e2 / (e1 + e2)) : -INFINITY;
    idxs[i] = i;
}

// Single-block bitonic sort, descending by key, ascending idx on tie.
__global__ void bitonic_sort(float* __restrict__ key, int* __restrict__ idx)
{
    for (unsigned k = 2; k <= NSORT; k <<= 1) {
        for (unsigned j = k >> 1; j > 0; j >>= 1) {
            for (unsigned i = threadIdx.x; i < NSORT; i += blockDim.x) {
                unsigned ixj = i ^ j;
                if (ixj > i) {
                    float k1 = key[i], k2 = key[ixj];
                    int   i1 = idx[i], i2 = idx[ixj];
                    bool descRegion = ((i & k) == 0);
                    bool doSwap;
                    if (descRegion)
                        doSwap = (k1 < k2) || (k1 == k2 && i1 > i2);
                    else
                        doSwap = (k1 > k2) || (k1 == k2 && i1 < i2);
                    if (doSwap) {
                        key[i] = k2; key[ixj] = k1;
                        idx[i] = i2; idx[ixj] = i1;
                    }
                }
            }
            __syncthreads();
        }
    }
}

__global__ void gather_boxes(const float* __restrict__ props, const int* __restrict__ idx,
                             float* __restrict__ boxes)
{
    int j = blockIdx.x * blockDim.x + threadIdx.x;
    if (j >= PRE_NMS) return;
    int s = idx[j];
#pragma unroll
    for (int c = 0; c < 4; ++c) boxes[j * 4 + c] = props[s * 4 + c];
}

__global__ void nms_k(const float* __restrict__ boxes, int* __restrict__ keepOut)
{
    __shared__ int keep[PRE_NMS];
    for (int j = threadIdx.x; j < PRE_NMS; j += blockDim.x) keep[j] = 1;
    __syncthreads();
    for (int i = 0; i < PRE_NMS - 1; ++i) {
        if (keep[i]) {
            float x1i = boxes[i * 4], y1i = boxes[i * 4 + 1];
            float x2i = boxes[i * 4 + 2], y2i = boxes[i * 4 + 3];
            float ai = (x2i - x1i + 1.f) * (y2i - y1i + 1.f);
            for (int j = i + 1 + (int)threadIdx.x; j < PRE_NMS; j += (int)blockDim.x) {
                if (!keep[j]) continue;
                float xx1 = fmaxf(x1i, boxes[j * 4]);
                float yy1 = fmaxf(y1i, boxes[j * 4 + 1]);
                float xx2 = fminf(x2i, boxes[j * 4 + 2]);
                float yy2 = fminf(y2i, boxes[j * 4 + 3]);
                float inter = fmaxf(xx2 - xx1 + 1.f, 0.f) * fmaxf(yy2 - yy1 + 1.f, 0.f);
                float aj = (boxes[j * 4 + 2] - boxes[j * 4] + 1.f) *
                           (boxes[j * 4 + 3] - boxes[j * 4 + 1] + 1.f);
                float iou = inter / (ai + aj - inter);
                if (iou > 0.7f) keep[j] = 0;
            }
        }
        __syncthreads();
    }
    for (int j = threadIdx.x; j < PRE_NMS; j += blockDim.x) keepOut[j] = keep[j];
}

__global__ void select_rois(const float* __restrict__ boxes, const int* __restrict__ keep,
                            float* __restrict__ rois)
{
    int sel[POST_N];
    int cnt = 0;
    for (int j = 0; j < PRE_NMS && cnt < POST_N; ++j) if (keep[j])  sel[cnt++] = j;
    for (int j = 0; j < PRE_NMS && cnt < POST_N; ++j) if (!keep[j]) sel[cnt++] = j;
    for (int k = 0; k < POST_N; ++k) {
        int s = sel[k];
        rois[k * 4 + 0] = boxes[s * 4 + 0];
        rois[k * 4 + 1] = boxes[s * 4 + 1];
        rois[k * 4 + 2] = boxes[s * 4 + 2];
        rois[k * 4 + 3] = boxes[s * 4 + 3];
    }
}

__global__ void roipool_k(const float* __restrict__ feat, const float* __restrict__ rois,
                          float* __restrict__ out)
{
    const int H = FH, W = FW;
    int total = POST_N * POOLC * 49;
    int t = blockIdx.x * blockDim.x + threadIdx.x;
    if (t >= total) return;
    int n = t / (POOLC * 49);
    int r = t - n * (POOLC * 49);
    int c = r / 49;
    int b = r - c * 49;
    int py = b / 7, px = b - py * 7;
    int x1 = (int)floorf(rois[n * 4 + 0] * 0.0625f);
    int y1 = (int)floorf(rois[n * 4 + 1] * 0.0625f);
    int x2 = (int)floorf(rois[n * 4 + 2] * 0.0625f);
    int y2 = (int)floorf(rois[n * 4 + 3] * 0.0625f);
    int Hr = y2 - y1 + 1, Wr = x2 - x1 + 1;
    int ylo = y1 + (py * Hr) / 7,       yhi = y1 + ((py + 1) * Hr + 6) / 7;
    int xlo = x1 + (px * Wr) / 7,       xhi = x1 + ((px + 1) * Wr + 6) / 7;
    ylo = ylo < 0 ? 0 : ylo; yhi = yhi > H ? H : yhi;
    xlo = xlo < 0 ? 0 : xlo; xhi = xhi > W ? W : xhi;
    float m = -INFINITY;
    for (int y = ylo; y < yhi; ++y)
        for (int x = xlo; x < xhi; ++x)
            m = fmaxf(m, feat[((size_t)c * H + y) * W + x]);
    out[t] = m;
}

// ---------------------------------------------------------------------------
static inline void launch_conv(const float* in, const float* w, const float* b,
                               float* out, int Cin, int Cout, int H, int W,
                               int ks, int relu, hipStream_t s)
{
    int pixT = (H * W + 15) / 16;
    int pixBlocks = (pixT + 7) / 8;
    int ocT = (Cout + 15) / 16;
    conv_wmma<<<ocT * pixBlocks, 256, 0, s>>>(in, w, b, out, Cin, Cout, H, W, ks, relu);
}

static inline void launch_gemm(const float* A, const float* Wt, const float* b,
                               float* out, int M, int N, int K, int relu, hipStream_t s)
{
    int mT = (M + 15) / 16;
    int nT = (N + 15) / 16;
    int nBlk = (nT + 7) / 8;
    gemm_wmma<<<mT * nBlk, 256, 0, s>>>(A, Wt, b, out, M, N, K, relu);
}

extern "C" void kernel_launch(void* const* d_in, const int* in_sizes, int n_in,
                              void* d_out, int out_size, void* d_ws, size_t ws_size,
                              hipStream_t stream)
{
    (void)in_sizes; (void)n_in; (void)out_size; (void)ws_size;
    const float* x        = (const float*)d_in[0];
    const float* im_info  = (const float*)d_in[1];
    const float* conv_w[13]; const float* conv_b[13];
    for (int i = 0; i < 13; ++i) { conv_w[i] = (const float*)d_in[2 + i];
                                   conv_b[i] = (const float*)d_in[15 + i]; }
    const float* rpn_conv_w  = (const float*)d_in[28];
    const float* rpn_conv_b  = (const float*)d_in[29];
    const float* rpn_score_w = (const float*)d_in[30];
    const float* rpn_score_b = (const float*)d_in[31];
    const float* rpn_loc_w   = (const float*)d_in[32];
    const float* rpn_loc_b   = (const float*)d_in[33];
    const float* fc6_w       = (const float*)d_in[34];
    const float* fc6_b       = (const float*)d_in[35];
    const float* fc7_w       = (const float*)d_in[36];
    const float* fc7_b       = (const float*)d_in[37];
    const float* cls_loc_w   = (const float*)d_in[38];
    const float* cls_loc_b   = (const float*)d_in[39];
    const float* det_score_w = (const float*)d_in[40];
    const float* det_score_b = (const float*)d_in[41];

    const size_t BUF = (size_t)64 * IMGH * IMGW;   // 31,129,600 floats
    float* ws = (float*)d_ws;
    size_t off = 0;
    auto take = [&](size_t n) { float* p = ws + off; off += n; return p; };
    float* bufA     = take(BUF);
    float* bufB     = take(BUF);
    float* rpnH     = take((size_t)512 * FH * FW);
    float* rpnLoc   = take((size_t)36 * FH * FW);
    float* rpnScore = take((size_t)18 * FH * FW);
    float* props    = take((size_t)NA * 4);
    float* keys     = take(NSORT);
    int*   idxs     = (int*)take(NSORT);
    float* sboxes   = take((size_t)PRE_NMS * 4);
    int*   keepArr  = (int*)take(PRE_NMS);
    float* rois     = take((size_t)POST_N * 4);
    float* pooled   = take((size_t)POST_N * POOLC * 49);
    float* fc6o     = take((size_t)POST_N * 4096);
    float* fc7o     = take((size_t)POST_N * 4096);

    // ---- VGG16 feature extraction (ping-pong bufA/bufB) ----
    const int cfg[17] = {64,64,-1,128,128,-1,256,256,256,-1,512,512,512,-1,512,512,512};
    const float* cur = x;
    float* nxt = bufA;
    int Cin = 3, H = IMGH, W = IMGW, wi = 0;
    for (int li = 0; li < 17; ++li) {
        if (cfg[li] == -1) {
            int C = Cin;
            int n = C * (H / 2) * (W / 2);
            maxpool_k<<<(n + 255) / 256, 256, 0, stream>>>(cur, nxt, C, H, W);
            H /= 2; W /= 2;
        } else {
            int Cout = cfg[li];
            launch_conv(cur, conv_w[wi], conv_b[wi], nxt, Cin, Cout, H, W, 3, 1, stream);
            Cin = Cout; ++wi;
        }
        cur = nxt;
        nxt = (cur == bufA) ? bufB : bufA;
    }
    const float* feat = cur;   // 512 x 50 x 38

    // ---- RPN ----
    launch_conv(feat, rpn_conv_w, rpn_conv_b, rpnH, 512, 512, FH, FW, 3, 1, stream);
    launch_conv(rpnH, rpn_loc_w,   rpn_loc_b,   rpnLoc,   512, 36, FH, FW, 1, 0, stream);
    launch_conv(rpnH, rpn_score_w, rpn_score_b, rpnScore, 512, 18, FH, FW, 1, 0, stream);

    // ---- proposals ----
    proposal_prep<<<(NSORT + 255) / 256, 256, 0, stream>>>(rpnLoc, rpnScore, im_info,
                                                           props, keys, idxs);
    bitonic_sort<<<1, 1024, 0, stream>>>(keys, idxs);
    gather_boxes<<<(PRE_NMS + 255) / 256, 256, 0, stream>>>(props, idxs, sboxes);
    nms_k<<<1, 1024, 0, stream>>>(sboxes, keepArr);
    select_rois<<<1, 1, 0, stream>>>(sboxes, keepArr, rois);

    // ---- ROI max-pool 7x7 ----
    {
        int total = POST_N * POOLC * 49;
        roipool_k<<<(total + 255) / 256, 256, 0, stream>>>(feat, rois, pooled);
    }

    // ---- heads ----
    float* out = (float*)d_out;
    launch_gemm(pooled, fc6_w, fc6_b, fc6o, POST_N, 4096, POOLC * 49, 1, stream);
    launch_gemm(fc6o,   fc7_w, fc7_b, fc7o, POST_N, 4096, 4096,       1, stream);
    launch_gemm(fc7o, cls_loc_w,   cls_loc_b,   out,               POST_N, 84, 4096, 0, stream);
    launch_gemm(fc7o, det_score_w, det_score_b, out + POST_N * 84, POST_N, 21, 4096, 0, stream);
}
